// RelativeMultiHeadAttentionBlock_1949915152890
// MI455X (gfx1250) — matrix operand order repaired and checked by reading, not compiled, not run on hardware
//
#include <hip/hip_runtime.h>
#include <stdint.h>

// ---------------- problem dimensions ----------------
#define B_      2
#define T_      1024
#define S_      2048
#define F_      1024
#define NHD_    1024      // N*H
#define HD_     64
#define NHEADS_ 16
#define MREL    2049      // S+1
#define MPAD    2064      // row stride for bd (16-aligned)
#define ATT_SCALE 0.125f  // 1/sqrt(64)

typedef uint16_t u16;
typedef __attribute__((ext_vector_type(4)))  uint32_t v4u;
typedef __attribute__((ext_vector_type(4)))  float    v4f;
typedef __attribute__((ext_vector_type(2)))  float    v2f;
typedef __attribute__((ext_vector_type(8)))  float    v8f;
typedef __attribute__((ext_vector_type(16))) __bf16   v16bf;

union BFrag { v16bf v; v4u q[2]; u16 s[16]; __bf16 h[16]; };

// native f32 -> bf16 (lets clang use v_cvt_*bf16* instead of ALU emulation)
static __device__ __forceinline__ u16 f2bf(float f) {
  union { __bf16 h; u16 u; } c; c.h = (__bf16)f;
  return c.u;
}
static __device__ __forceinline__ float bf2f(u16 h) {
  union { uint32_t u; float f; } c; c.u = ((uint32_t)h) << 16;
  return c.f;
}
static __device__ __forceinline__ v8f wmma_bf16(v16bf a, v16bf b, v8f c) {
  return __builtin_amdgcn_wmma_f32_16x16x32_bf16(false, a, false, b, (short)0, c,
                                                 false, false);
}
#define V8F_ZERO ((v8f){0.f,0.f,0.f,0.f,0.f,0.f,0.f,0.f})

// =====================================================================
// Kernel 1: C(MxN) = A(MxK,f32) x W(KxN,f32) -> bf16.
// Block 128 thr = 4 waves; block tile 128x32; wave tile 32x32
// (2 row sub-tiles x 2 col tiles = 4 WMMAs / K-step, B-frags reused).
// W tile staged transposed in LDS (coalesced f32 reads, ds_load_b128 frags).
// DUAL: write out0=C+bias0, out1=C+bias1 (q_u / q_v).
// GUARD: M not tile-divisible (r). TMODE==1: store (b, col, s) (v^T).
// =====================================================================
template <bool DUAL, bool GUARD, int TMODE>
__global__ __launch_bounds__(128) void gemm_proj(
    const float* __restrict__ A, const float* __restrict__ W,
    const float* __restrict__ bias0, const float* __restrict__ bias1,
    u16* __restrict__ out0, u16* __restrict__ out1,
    int Mrows, int K, int Ncols, int Srows)
{
  __shared__ __attribute__((aligned(16))) u16 wt[32][40];  // [n][k], padded

  const int tid  = threadIdx.x;
  const int lane = tid & 31, wave = tid >> 5;
  const int half = lane >> 4, l16 = lane & 15;

  const int n0  = blockIdx.x * 32;
  const int m0w = blockIdx.y * 128 + wave * 32;

  int r0 = m0w + l16, r1 = m0w + 16 + l16;
  if (GUARD) {
    r0 = r0 < Mrows ? r0 : (Mrows - 1);
    r1 = r1 < Mrows ? r1 : (Mrows - 1);
  }
  const float* Arow0 = A + (size_t)r0 * K;
  const float* Arow1 = A + (size_t)r1 * K;

  v8f acc00 = V8F_ZERO, acc01 = V8F_ZERO, acc10 = V8F_ZERO, acc11 = V8F_ZERO;

  for (int k0 = 0; k0 < K; k0 += 32) {
    // ---- cooperative W tile -> LDS (transposed, bf16) ----
    #pragma unroll
    for (int i = 0; i < 8; ++i) {
      int flat = i * 128 + tid;
      int kk = flat >> 5, nn = flat & 31;
      wt[nn][kk] = f2bf(W[(size_t)(k0 + kk) * Ncols + n0 + nn]);
    }
    __syncthreads();

    // ---- A fragments: e<8 -> K=half*8+e ; e>=8 -> K=16+half*8+(e-8) ----
    BFrag a0, a1;
    {
      const float* p = Arow0 + k0 + half * 8;
      v4f x0 = *(const v4f*)p, x1 = *(const v4f*)(p + 4);
      v4f x2 = *(const v4f*)(p + 16), x3 = *(const v4f*)(p + 20);
      const float* q = Arow1 + k0 + half * 8;
      v4f y0 = *(const v4f*)q, y1 = *(const v4f*)(q + 4);
      v4f y2 = *(const v4f*)(q + 16), y3 = *(const v4f*)(q + 20);
      #pragma unroll
      for (int i = 0; i < 4; ++i) {
        a0.h[i] = (__bf16)x0[i]; a0.h[4+i] = (__bf16)x1[i];
        a0.h[8+i] = (__bf16)x2[i]; a0.h[12+i] = (__bf16)x3[i];
        a1.h[i] = (__bf16)y0[i]; a1.h[4+i] = (__bf16)y1[i];
        a1.h[8+i] = (__bf16)y2[i]; a1.h[12+i] = (__bf16)y3[i];
      }
    }

    // ---- B fragments (col = lane%16, e<8 -> K=half*8+e ; e>=8 -> K=16+half*8+(e-8)) + 4 WMMAs ----
    BFrag b0, b1;
    { const u16* bp = &wt[l16][half * 8];
      b0.q[0] = *(const v4u*)bp; b0.q[1] = *(const v4u*)(bp + 16); }
    { const u16* bp = &wt[16 + l16][half * 8];
      b1.q[0] = *(const v4u*)bp; b1.q[1] = *(const v4u*)(bp + 16); }
    acc00 = wmma_bf16(a0.v, b0.v, acc00);
    acc01 = wmma_bf16(a0.v, b1.v, acc01);
    acc10 = wmma_bf16(a1.v, b0.v, acc10);
    acc11 = wmma_bf16(a1.v, b1.v, acc11);
    __syncthreads();
  }

  // ---- epilogue: C layout row = e + 8*half, col = lane%16 ----
  #pragma unroll
  for (int rt = 0; rt < 2; ++rt) {
    #pragma unroll
    for (int ct = 0; ct < 2; ++ct) {
      const int colg = n0 + ct * 16 + l16;
      const v8f& acc = rt == 0 ? (ct == 0 ? acc00 : acc01)
                               : (ct == 0 ? acc10 : acc11);
      float b0 = 0.f, b1 = 0.f;
      if (DUAL) { b0 = bias0[colg]; b1 = bias1[colg]; }
      #pragma unroll
      for (int e = 0; e < 8; ++e) {
        int row = m0w + rt * 16 + e + 8 * half;
        if (!GUARD || row < Mrows) {
          float c = acc[e];
          size_t idx;
          if (TMODE == 0) {
            idx = (size_t)row * Ncols + colg;
          } else {                     // store transposed per batch: (b, col, s)
            int bb = row / Srows, ss = row - bb * Srows;
            idx = ((size_t)(bb * Ncols + colg)) * Srows + ss;
          }
          out0[idx] = f2bf(c + b0);
          if (DUAL) out1[idx] = f2bf(c + b1);
        }
      }
    }
  }
}

// =====================================================================
// Kernel 2: bd[b,n,t,m] = sum_h q_v[b,t,n,h] * r[m,n,h]   (K=64 -> 2 WMMAs)
// grid = (ceil(129/4), T/16, B*N), block = 128 (4 waves, wave = one m-tile)
// =====================================================================
__global__ __launch_bounds__(128) void bd_gemm(
    const u16* __restrict__ qv, const u16* __restrict__ rb, u16* __restrict__ bd)
{
  const int tid  = threadIdx.x;
  const int lane = tid & 31, wave = tid >> 5;
  const int half = lane >> 4, l16 = lane & 15;

  const int m0 = (blockIdx.x * 4 + wave) * 16;
  if (m0 >= MREL) return;                       // wave-uniform
  const int t0 = blockIdx.y * 16;
  const int bn = blockIdx.z;                    // b*16 + n
  const int b = bn >> 4, n = bn & 15;

  const u16* qrow = qv + ((size_t)(b * T_ + t0 + l16)) * NHD_ + n * HD_;
  const int  mcl  = (m0 + l16) < (MREL - 1) ? (m0 + l16) : (MREL - 1);
  const u16* rrow = rb + (size_t)mcl * NHD_ + n * HD_;

  v8f acc = V8F_ZERO;
  #pragma unroll
  for (int ks = 0; ks < 2; ++ks) {
    BFrag af;
    const u16* ap = qrow + ks * 32;
    af.q[0] = *(const v4u*)(ap + half * 8);
    af.q[1] = *(const v4u*)(ap + 16 + half * 8);
    BFrag bf;
    const u16* bp = rrow + ks * 32 + half * 8;
    bf.q[0] = *(const v4u*)bp; bf.q[1] = *(const v4u*)(bp + 16);
    acc = wmma_bf16(af.v, bf.v, acc);
  }
  #pragma unroll
  for (int e = 0; e < 8; ++e) {
    int t = t0 + e + 8 * half;
    int m = m0 + l16;
    if (m < MREL) bd[((size_t)bn * T_ + t) * MPAD + m] = f2bf(acc[e]);
  }
}

// =====================================================================
// Kernel 3: flash attention per (b,n, 16-row q tile); online softmax.
// k-tiles (shared by all 4 waves of the block) are double-buffered in LDS
// via GLOBAL_LOAD_ASYNC_TO_LDS_B128 (ASYNCcnt) and drained with
// s_wait_asynccnt, overlapping the next tile's fetch with current WMMAs.
// Relative-shift bias gathered from bd:
//   d=s-t:  d<=1025 -> bd[t, d+1023] ; d==1026 -> 0 ; d>=1027 -> bd[t+1, d-1027]
// =====================================================================
__global__ __launch_bounds__(128) void attention(
    const u16* __restrict__ qu, const u16* __restrict__ kb,
    const u16* __restrict__ vbt, const u16* __restrict__ bd,
    float* __restrict__ attn)
{
  __shared__ __attribute__((aligned(16))) u16 kbuf[2][32][64];  // 8 KB
  __shared__ __attribute__((aligned(16))) u16 plds[4][16][40];  // 5 KB

  const int tid  = threadIdx.x;
  const int lane = tid & 31, wave = tid >> 5;
  const int half = lane >> 4, l16 = lane & 15;

  const int t0 = blockIdx.x * 64 + wave * 16;
  const int bn = blockIdx.y;
  const int b = bn >> 4, n = bn & 15;

  const u16* kbase  = kb  + ((size_t)b * S_) * NHD_ + n * HD_;
  const u16* vbase  = vbt + ((size_t)(b * NHD_ + n * HD_)) * S_;
  const u16* bdbase = bd  + (size_t)bn * T_ * MPAD;

  // async-stage one 32x64 bf16 k tile (4 KB = 256 x 16B, 2 chunks/thread)
  auto issue_ktile = [&](int buf, int sb) {
    #pragma unroll
    for (int i = 0; i < 2; ++i) {
      int c = tid + i * 128;                   // chunk 0..255
      int row = c >> 3, off = (c & 7) * 16;    // 8 x 16B per 128B row
      uint64_t ga = (uint64_t)(uintptr_t)(kbase + (size_t)(sb + row) * NHD_) + off;
      uint32_t la = (uint32_t)(uintptr_t)(&kbuf[buf][row][0]) + off;
      asm volatile("global_load_async_to_lds_b128 %0, %1, off"
                   :: "v"(la), "v"(ga) : "memory");
    }
  };

  // preload q_u A-fragments (2 K-steps over H=64)
  const u16* qrow = qu + ((size_t)(b * T_ + t0 + l16)) * NHD_ + n * HD_;
  BFrag qa[2];
  #pragma unroll
  for (int ks = 0; ks < 2; ++ks) {
    const u16* ap = qrow + ks * 32;
    qa[ks].q[0] = *(const v4u*)(ap + half * 8);
    qa[ks].q[1] = *(const v4u*)(ap + 16 + half * 8);
  }

  v8f O[4];
  float mrow[8], lrow[8];
  #pragma unroll
  for (int ht = 0; ht < 4; ++ht) O[ht] = V8F_ZERO;
  #pragma unroll
  for (int e = 0; e < 8; ++e) { mrow[e] = -__builtin_inff(); lrow[e] = 0.f; }

  issue_ktile(0, 0);

  for (int sb = 0; sb < S_; sb += 32) {
    const int cur = (sb >> 5) & 1;
    asm volatile("s_wait_asynccnt 0" ::: "memory");  // own chunks landed
    __syncthreads();                                  // all waves' chunks landed
    if (sb + 32 < S_) issue_ktile(cur ^ 1, sb + 32);  // overlap next fetch

    // prefetch v tile for this s-block (global_prefetch)
    __builtin_prefetch(vbase + ((size_t)l16) * S_ + sb, 0, 3);

    float lv[2][8];
    #pragma unroll
    for (int j = 0; j < 2; ++j) {
      v8f lacc = V8F_ZERO;
      const u16* kr = &kbuf[cur][j * 16 + l16][0];
      #pragma unroll
      for (int ks = 0; ks < 2; ++ks) {
        BFrag bf;
        const u16* bp = kr + ks * 32 + half * 8;
        bf.q[0] = *(const v4u*)bp; bf.q[1] = *(const v4u*)(bp + 16);
        lacc = wmma_bf16(qa[ks].v, bf.v, lacc);
      }
      #pragma unroll
      for (int e = 0; e < 8; ++e) {
        int t = t0 + e + 8 * half;
        int s = sb + j * 16 + l16;
        int d = s - t;
        float bias;
        if (d == 1026) {
          bias = 0.f;
        } else {
          int tt = (d <= 1025) ? t : (t + 1);
          int m  = (d <= 1025) ? (d + 1023) : (d - 1027);
          bias = bf2f(bdbase[(size_t)tt * MPAD + m]);
        }
        lv[j][e] = (lacc[e] + bias) * ATT_SCALE;
      }
    }

    // ---- online softmax (row stats shared by the 16 lanes of each half) ----
    #pragma unroll
    for (int e = 0; e < 8; ++e) {
      float rm = fmaxf(lv[0][e], lv[1][e]);
      rm = fmaxf(rm, __shfl_xor(rm, 1, 32));
      rm = fmaxf(rm, __shfl_xor(rm, 2, 32));
      rm = fmaxf(rm, __shfl_xor(rm, 4, 32));
      rm = fmaxf(rm, __shfl_xor(rm, 8, 32));
      float nm    = fmaxf(mrow[e], rm);
      float alpha = __expf(mrow[e] - nm);
      mrow[e] = nm;
      float p0 = __expf(lv[0][e] - nm);
      float p1 = __expf(lv[1][e] - nm);
      float rs = p0 + p1;
      rs += __shfl_xor(rs, 1, 32);
      rs += __shfl_xor(rs, 2, 32);
      rs += __shfl_xor(rs, 4, 32);
      rs += __shfl_xor(rs, 8, 32);
      lrow[e] = lrow[e] * alpha + rs;
      #pragma unroll
      for (int ht = 0; ht < 4; ++ht) O[ht][e] *= alpha;
      int r = e + 8 * half;
      plds[wave][r][l16]      = f2bf(p0);
      plds[wave][r][16 + l16] = f2bf(p1);
    }
    // DS ops are in-order within a wave; keep the compiler from reordering.
    asm volatile("" ::: "memory");

    // probs A-fragment from LDS (16x32, K = s within block)
    BFrag pa;
    const u16* pr = &plds[wave][l16][0];
    pa.q[0] = *(const v4u*)(pr + half * 8);
    pa.q[1] = *(const v4u*)(pr + 16 + half * 8);

    // PV: v stored transposed (B,N,H,S) -> B-frag is two 16B chunks per lane (K=half*8, 16+half*8)
    #pragma unroll
    for (int ht = 0; ht < 4; ++ht) {
      const u16* vp = vbase + ((size_t)(ht * 16 + l16)) * S_ + sb + half * 8;
      BFrag vf; vf.q[0] = *(const v4u*)vp; vf.q[1] = *(const v4u*)(vp + 16);
      O[ht] = wmma_bf16(pa.v, vf.v, O[ht]);
    }
    __syncthreads();   // all reads of kbuf[cur] done before it is re-filled
  }

  // ---- normalize + store f32 attention output (B,T,N*H) ----
  #pragma unroll
  for (int e = 0; e < 8; ++e) {
    float inv = 1.0f / lrow[e];
    int t = t0 + e + 8 * half;
    float* orow = attn + ((size_t)(b * T_ + t)) * NHD_ + n * HD_;
    #pragma unroll
    for (int ht = 0; ht < 4; ++ht) orow[ht * 16 + l16] = O[ht][e] * inv;
  }
}

// =====================================================================
// Kernel 4: out = attn(2048x1024,f32) x Wout(1024x1024,f32) + b_out
// Uses V_WMMA_F32_16X16X4_F32 (keeps reference f32 precision).
// =====================================================================
#if defined(__AMDGCN__) && __has_builtin(__builtin_amdgcn_wmma_f32_16x16x4_f32)
#define HAVE_WMMA_F32 1
#else
#define HAVE_WMMA_F32 0
#endif

__global__ __launch_bounds__(128) void out_proj(
    const float* __restrict__ attn, const float* __restrict__ Wout,
    const float* __restrict__ bout, float* __restrict__ out)
{
  const int tid  = threadIdx.x;
  const int lane = tid & 31, wave = tid >> 5;
  const int half = lane >> 4, l16 = lane & 15;
  const int colg = blockIdx.x * 16 + l16;
  const int row0 = blockIdx.y * 64 + wave * 16;

#if HAVE_WMMA_F32
  v8f acc = V8F_ZERO;
  const float* arow = attn + (size_t)(row0 + l16) * NHD_;
  for (int k0 = 0; k0 < F_; k0 += 4) {
    // A 16x4: element e -> K = k0 + 2*half + e (contiguous pair)
    v2f a = *(const v2f*)(arow + k0 + 2 * half);
    v2f bv;
    bv[0] = Wout[(size_t)(k0 + 2 * half)     * F_ + colg];
    bv[1] = Wout[(size_t)(k0 + 2 * half + 1) * F_ + colg];
    acc = __builtin_amdgcn_wmma_f32_16x16x4_f32(false, a, false, bv, (short)0,
                                                acc, false, false);
  }
  const float bb = bout[colg];
  #pragma unroll
  for (int e = 0; e < 8; ++e)
    out[(size_t)(row0 + e + 8 * half) * F_ + colg] = acc[e] + bb;
#else
  const float bb = bout[colg];
  #pragma unroll
  for (int e = 0; e < 8; ++e) {
    int row = row0 + e + 8 * half;
    float s = 0.f;
    for (int k = 0; k < F_; ++k)
      s += attn[(size_t)row * NHD_ + k] * Wout[(size_t)k * F_ + colg];
    out[(size_t)row * F_ + colg] = s + bb;
  }
#endif
}

// =====================================================================
// Host launcher
// =====================================================================
extern "C" void kernel_launch(void* const* d_in, const int* in_sizes, int n_in,
                              void* d_out, int out_size, void* d_ws, size_t ws_size,
                              hipStream_t stream) {
  const float* x    = (const float*)d_in[0];
  const float* rel  = (const float*)d_in[1];
  const float* mem  = (const float*)d_in[2];
  const float* Wq   = (const float*)d_in[3];
  const float* Wk   = (const float*)d_in[4];
  const float* Wv   = (const float*)d_in[5];
  const float* Wr   = (const float*)d_in[6];
  const float* ub   = (const float*)d_in[7];
  const float* vb   = (const float*)d_in[8];
  const float* Wout = (const float*)d_in[9];
  const float* bout = (const float*)d_in[10];
  float* out = (float*)d_out;

  char* ws = (char*)d_ws;
  size_t off = 0;
  auto take = [&](size_t bytes) {
    char* p = ws + off;
    off += (bytes + 255) & ~(size_t)255;
    return p;
  };
  u16*   qu   = (u16*)  take((size_t)B_ * T_ * NHD_ * 2);           // 4 MB
  u16*   qv   = (u16*)  take((size_t)B_ * T_ * NHD_ * 2);           // 4 MB
  u16*   kbf  = (u16*)  take((size_t)B_ * S_ * NHD_ * 2);           // 8 MB
  u16*   vbt  = (u16*)  take((size_t)B_ * NHD_ * S_ * 2);           // 8 MB (v^T)
  u16*   rbf  = (u16*)  take((size_t)MREL * NHD_ * 2);              // ~4 MB
  u16*   bdws = (u16*)  take((size_t)B_ * NHEADS_ * T_ * MPAD * 2); // ~135 MB
  float* attn = (float*)take((size_t)B_ * T_ * NHD_ * 4);           // 8 MB

  const dim3 blk(128);

  // q (+u_bias -> q_u, +v_bias -> q_v), M = B*T = 2048
  gemm_proj<true, false, 0><<<dim3(32, 16), blk, 0, stream>>>(
      x, Wq, ub, vb, qu, qv, B_ * T_, F_, NHD_, 0);
  // k, M = B*S = 4096
  gemm_proj<false, false, 0><<<dim3(32, 32), blk, 0, stream>>>(
      mem, Wk, nullptr, nullptr, kbf, nullptr, B_ * S_, F_, NHD_, 0);
  // v (stored transposed (B,N,H,S) for PV B-fragments)
  gemm_proj<false, false, 1><<<dim3(32, 32), blk, 0, stream>>>(
      mem, Wv, nullptr, nullptr, vbt, nullptr, B_ * S_, F_, NHD_, S_);
  // r, M = 2049 (tail guarded)
  gemm_proj<false, true, 0><<<dim3(32, 17), blk, 0, stream>>>(
      rel, Wr, nullptr, nullptr, rbf, nullptr, MREL, F_, NHD_, 0);
  // bd = q_v . r^T  per (b,n)
  bd_gemm<<<dim3(33, T_ / 16, B_ * NHEADS_), blk, 0, stream>>>(qv, rbf, bdws);
  // fused relative-shift bias + flash attention (async-LDS k staging)
  attention<<<dim3(T_ / 64, B_ * NHEADS_), blk, 0, stream>>>(qu, kbf, vbt, bdws, attn);
  // final projection (f32 WMMA)
  out_proj<<<dim3(F_ / 16, (B_ * T_) / 64), blk, 0, stream>>>(attn, Wout, bout, out);
}
